// qGCN_26414048870739
// MI455X (gfx1250) — 16-bit and 8-bit matrix operand order repaired and checked by reading, compiled, not run on hardware
//
#include <hip/hip_runtime.h>
#include <stdint.h>

typedef __attribute__((ext_vector_type(16))) _Float16 v16h;
typedef __attribute__((ext_vector_type(8)))  _Float16 v8h;
typedef __attribute__((ext_vector_type(8)))  float    v8f;
typedef __attribute__((ext_vector_type(8)))  int      v8i;
typedef __attribute__((ext_vector_type(4)))  int      v4i;

union AFragH { v16h v; v8h p[2]; };
union AFrag8 { v8i v; uint64_t q[4]; };
union BFrag8 { v8i v; v4i p[2]; };

__global__ void k_deg_init(float* deg, int N) {
  int i = blockIdx.x * blockDim.x + threadIdx.x;
  if (i < N) deg[i] = 1.0f;
}
__global__ void k_deg_count(const int* __restrict__ dst, float* deg, int E) {
  int e = blockIdx.x * blockDim.x + threadIdx.x;
  if (e < E) atomicAdd(&deg[dst[e]], 1.0f);
}
__global__ void k_dinv(float* deg, int N) {
  int i = blockIdx.x * blockDim.x + threadIdx.x;
  if (i < N) deg[i] = rsqrtf(deg[i]);
}

__global__ void k_split_x(const float* __restrict__ x, _Float16* __restrict__ xh,
                          _Float16* __restrict__ xl, int total) {
  int i = blockIdx.x * blockDim.x + threadIdx.x;
  if (i >= total) return;
  float v = x[i];
  _Float16 h = (_Float16)v;
  xh[i] = h;
  xl[i] = (_Float16)(v - (float)h);
}

__global__ void k_quant_w_f16(const float* __restrict__ W, const float* a_s,
                              _Float16* __restrict__ Wt, int fin, int fout) {
  int i = blockIdx.x * blockDim.x + threadIdx.x;
  if (i >= fin * fout) return;
  int k = i / fout, n = i % fout;
  float q = rintf(fminf(fmaxf(W[i] / a_s[0], -8.0f), 7.0f));
  Wt[(size_t)n * fin + k] = (_Float16)q;
}
__global__ void k_quant_w_i8(const float* __restrict__ W, const float* a_s,
                             signed char* __restrict__ Wt, int fin, int fout, int fout_pad) {
  int i = blockIdx.x * blockDim.x + threadIdx.x;
  if (i >= fin * fout_pad) return;
  int k = i / fout_pad, n = i % fout_pad;
  float q = 0.0f;
  if (n < fout) q = rintf(fminf(fmaxf(W[(size_t)k * fout + n] / a_s[0], -8.0f), 7.0f));
  Wt[(size_t)n * fin + k] = (signed char)q;
}

__global__ void k_gemm_f16_split(const _Float16* __restrict__ Ahi, const _Float16* __restrict__ Alo,
                                 const _Float16* __restrict__ Bt, const float* __restrict__ bias,
                                 const float* __restrict__ a_s, float* __restrict__ OUT, int N) {
  int lane = threadIdx.x & 31;
  int wave = threadIdx.x >> 5;
  int half = lane >> 4;
  int row  = blockIdx.x * 16 + (lane & 15);
  if (row > N - 1) row = N - 1;
  int col  = wave * 16 + (lane & 15);
  const _Float16* ah = Ahi + (size_t)row * 128;
  const _Float16* al = Alo + (size_t)row * 128;
  const _Float16* br = Bt  + (size_t)col * 128;
  v8f acc = {};
#pragma unroll
  for (int k0 = 0; k0 < 128; k0 += 32) {
    AFragH fh, fl;
    fh.p[0] = *(const v8h*)(ah + k0 + half * 8);
    fh.p[1] = *(const v8h*)(ah + k0 + 16 + half * 8);
    fl.p[0] = *(const v8h*)(al + k0 + half * 8);
    fl.p[1] = *(const v8h*)(al + k0 + 16 + half * 8);
    AFragH fb; fb.p[0] = *(const v8h*)(br + k0 + half * 8); fb.p[1] = *(const v8h*)(br + k0 + 16 + half * 8);
    v16h b = fb.v;
    acc = __builtin_amdgcn_wmma_f32_16x16x32_f16(false, fh.v, false, b, (short)0, acc, false, false);
    acc = __builtin_amdgcn_wmma_f32_16x16x32_f16(false, fl.v, false, b, (short)0, acc, false, false);
  }
  float s  = a_s[0];
  float bc = bias[col];
#pragma unroll
  for (int r = 0; r < 8; ++r) {
    int rr = blockIdx.x * 16 + half * 8 + r;
    if (rr < N) OUT[(size_t)rr * 128 + col] = s * acc[r] + bc;
  }
}

__global__ void k_gemm_iu8(const unsigned char* __restrict__ A, const signed char* __restrict__ Bt,
                           const float* __restrict__ bias, const float* __restrict__ a_s,
                           const float* __restrict__ grow, float* __restrict__ OUT,
                           int N, int Cout, int ldout) {
  int lane = threadIdx.x & 31;
  int wave = threadIdx.x >> 5;
  int half = lane >> 4;
  int row  = blockIdx.x * 16 + (lane & 15);
  if (row > N - 1) row = N - 1;
  int col  = wave * 16 + (lane & 15);
  const unsigned char* ar = A  + (size_t)row * 128;
  const signed char*   br = Bt + (size_t)col * 128;
  v8i acc = {};
#pragma unroll
  for (int k0 = 0; k0 < 128; k0 += 64) {
    AFrag8 fa;
    fa.q[0] = *(const uint64_t*)(ar + k0 +      half * 16);
    fa.q[1] = *(const uint64_t*)(ar + k0 + 8  + half * 16);
    fa.q[2] = *(const uint64_t*)(ar + k0 + 32 + half * 16);
    fa.q[3] = *(const uint64_t*)(ar + k0 + 40 + half * 16);
    BFrag8 fb;
    fb.p[0] = *(const v4i*)(br + k0 +      half * 16);
    fb.p[1] = *(const v4i*)(br + k0 + 32 + half * 16);
    acc = __builtin_amdgcn_wmma_i32_16x16x64_iu8(false, fa.v, true, fb.v, acc, false, false);
  }
  if (col < Cout) {
    float s  = a_s[0];
    float bc = bias[col];
#pragma unroll
    for (int r = 0; r < 8; ++r) {
      int rr = blockIdx.x * 16 + half * 8 + r;
      if (rr < N) OUT[(size_t)rr * ldout + col] = grow[rr] * s * (float)acc[r] + bc;
    }
  }
}

__global__ void k_agg_init(const float* __restrict__ H, float* __restrict__ OUT,
                           const float* __restrict__ dinv, int N, int C) {
  int i = blockIdx.x * blockDim.x + threadIdx.x;
  if (i >= N * C) return;
  int r = i / C;
  float d = dinv[r];
  OUT[i] = H[i] * d * d;
}
__global__ void k_edge_scatter(const float* __restrict__ H, float* __restrict__ OUT,
                               const int* __restrict__ src, const int* __restrict__ dst,
                               const float* __restrict__ dinv, int E, int C) {
  int gwave  = (blockIdx.x * blockDim.x + threadIdx.x) >> 5;
  int lane   = threadIdx.x & 31;
  int nwaves = (gridDim.x * blockDim.x) >> 5;
  for (int e = gwave; e < E; e += nwaves) {
    int en = e + nwaves;
    if (en < E) __builtin_prefetch(H + (size_t)src[en] * C, 0, 1);
    int s = src[e], d = dst[e];
    float w = dinv[s] * dinv[d];
    const float* hs = H   + (size_t)s * C;
    float*       od = OUT + (size_t)d * C;
    for (int c = lane; c < C; c += 32)
      atomicAdd(&od[c], hs[c] * w);
  }
}

__global__ void k_bn_zero(float* sums) { if (threadIdx.x < 256) sums[threadIdx.x] = 0.0f; }
__global__ void k_bn_stats(const float* __restrict__ X, float* __restrict__ sums, int N) {
  int ch = threadIdx.x;
  int r0 = blockIdx.x * 256;
  int r1 = r0 + 256; if (r1 > N) r1 = N;
  float s = 0.0f, s2 = 0.0f;
  for (int r = r0; r < r1; ++r) { float v = X[(size_t)r * 128 + ch]; s += v; s2 += v * v; }
  atomicAdd(&sums[ch], s);
  atomicAdd(&sums[128 + ch], s2);
}
__global__ void k_bn_final(const float* __restrict__ sums, const float* __restrict__ g,
                           const float* __restrict__ b, float* __restrict__ ss, int N) {
  int c = threadIdx.x;
  float mean = sums[c] / (float)N;
  float var  = fmaxf(sums[128 + c] / (float)N - mean * mean, 0.0f);
  float sc   = g[c] * rsqrtf(var + 1e-5f);
  ss[c] = sc;
  ss[128 + c] = b[c] - mean * sc;
}
__global__ void k_bn_relu_quant(const float* __restrict__ X, const float* __restrict__ ss,
                                const float* __restrict__ grow, unsigned char* __restrict__ codes,
                                int N) {
  int i = blockIdx.x * blockDim.x + threadIdx.x;
  if (i >= N * 128) return;
  int r = i >> 7, c = i & 127;
  float v = fmaxf(ss[c] * X[i] + ss[128 + c], 0.0f);
  float q = rintf(fminf(v / grow[r], 15.0f));
  codes[i] = (unsigned char)q;
}

__global__ void k_logsoftmax(const float* __restrict__ H, float* __restrict__ OUT, int N, int C) {
  int row  = (blockIdx.x * blockDim.x + threadIdx.x) >> 5;
  int lane = threadIdx.x & 31;
  if (row >= N) return;
  const float* h = H + (size_t)row * C;
  float v0 = h[lane];
  float v1 = (lane + 32 < C) ? h[lane + 32] : -3.4e38f;
  float m = fmaxf(v0, v1);
  for (int o = 16; o; o >>= 1) m = fmaxf(m, __shfl_xor(m, o, 32));
  float s = expf(v0 - m) + ((lane + 32 < C) ? expf(v1 - m) : 0.0f);
  for (int o = 16; o; o >>= 1) s += __shfl_xor(s, o, 32);
  float lse = m + logf(s);
  OUT[(size_t)row * C + lane] = v0 - lse;
  if (lane + 32 < C) OUT[(size_t)row * C + lane + 32] = v1 - lse;
}

extern "C" void kernel_launch(void* const* d_in, const int* in_sizes, int n_in,
                              void* d_out, int out_size, void* d_ws, size_t ws_size,
                              hipStream_t stream) {
  const float* x    = (const float*)d_in[0];
  const int*   ei   = (const int*)d_in[1];
  const float* W1   = (const float*)d_in[2];
  const float* b1   = (const float*)d_in[3];
  const float* a1   = (const float*)d_in[4];
  const float* W2   = (const float*)d_in[5];
  const float* b2   = (const float*)d_in[6];
  const float* a2   = (const float*)d_in[7];
  const float* g2   = (const float*)d_in[8];
  const float* W3   = (const float*)d_in[9];
  const float* b3   = (const float*)d_in[10];
  const float* a3   = (const float*)d_in[11];
  const float* g3   = (const float*)d_in[12];
  const float* bn1g = (const float*)d_in[13];
  const float* bn1b = (const float*)d_in[14];
  const float* bn2g = (const float*)d_in[15];
  const float* bn2b = (const float*)d_in[16];

  const int N = in_sizes[0] / 128;
  const int E = in_sizes[1] / 2;
  const int C = in_sizes[10];
  const int Cpad = ((C + 15) / 16) * 16;
  const int* src = ei;
  const int* dst = ei + E;

  char*  base = (char*)d_ws;
  size_t off  = 0;
  auto carve = [&](size_t bytes) -> void* {
    void* p = base + off;
    off += (bytes + 255) & ~(size_t)255;
    return p;
  };
  float*        dinv  = (float*)carve((size_t)N * 4);
  float*        bufA  = (float*)carve((size_t)N * 128 * 4);
  float*        bufB  = (float*)carve((size_t)N * 128 * 4);
  _Float16*     xhi   = (_Float16*)carve((size_t)N * 128 * 2);
  _Float16*     xlo   = (_Float16*)carve((size_t)N * 128 * 2);
  unsigned char* codes = (unsigned char*)xhi;
  _Float16*     Wt1   = (_Float16*)carve(128 * 128 * 2);
  signed char*  Wt2   = (signed char*)carve(128 * 128);
  signed char*  Wt3   = (signed char*)carve((size_t)Cpad * 128);
  float*        bnsum = (float*)carve(256 * 4);
  float*        bnss  = (float*)carve(256 * 4);

  const int rowtiles = (N + 15) / 16;
  const int totF     = N * 128;
  const int gF       = (totF + 255) / 256;

  k_deg_init<<<(N + 255) / 256, 256, 0, stream>>>(dinv, N);
  k_deg_count<<<(E + 255) / 256, 256, 0, stream>>>(dst, dinv, E);
  k_dinv<<<(N + 255) / 256, 256, 0, stream>>>(dinv, N);
  k_split_x<<<gF, 256, 0, stream>>>(x, xhi, xlo, totF);
  k_quant_w_f16<<<(128 * 128 + 255) / 256, 256, 0, stream>>>(W1, a1, Wt1, 128, 128);
  k_quant_w_i8<<<(128 * 128 + 255) / 256, 256, 0, stream>>>(W2, a2, Wt2, 128, 128, 128);
  k_quant_w_i8<<<(128 * Cpad + 255) / 256, 256, 0, stream>>>(W3, a3, Wt3, 128, C, Cpad);

  k_gemm_f16_split<<<rowtiles, 256, 0, stream>>>(xhi, xlo, Wt1, b1, a1, bufA, N);
  k_agg_init<<<gF, 256, 0, stream>>>(bufA, bufB, dinv, N, 128);
  k_edge_scatter<<<1024, 256, 0, stream>>>(bufA, bufB, src, dst, dinv, E, 128);
  k_bn_zero<<<1, 256, 0, stream>>>(bnsum);
  k_bn_stats<<<(N + 255) / 256, 128, 0, stream>>>(bufB, bnsum, N);
  k_bn_final<<<1, 128, 0, stream>>>(bnsum, bn1g, bn1b, bnss, N);
  k_bn_relu_quant<<<gF, 256, 0, stream>>>(bufB, bnss, g2, codes, N);

  k_gemm_iu8<<<rowtiles, 256, 0, stream>>>(codes, Wt2, b2, a2, g2, bufA, N, 128, 128);
  k_agg_init<<<gF, 256, 0, stream>>>(bufA, bufB, dinv, N, 128);
  k_edge_scatter<<<1024, 256, 0, stream>>>(bufA, bufB, src, dst, dinv, E, 128);
  k_bn_zero<<<1, 256, 0, stream>>>(bnsum);
  k_bn_stats<<<(N + 255) / 256, 128, 0, stream>>>(bufB, bnsum, N);
  k_bn_final<<<1, 128, 0, stream>>>(bnsum, bn2g, bn2b, bnss, N);
  k_bn_relu_quant<<<gF, 256, 0, stream>>>(bufB, bnss, g3, codes, N);

  k_gemm_iu8<<<rowtiles, 96, 0, stream>>>(codes, Wt3, b3, a3, g3, bufA, N, C, C);
  const int totC = N * C;
  k_agg_init<<<(totC + 255) / 256, 256, 0, stream>>>(bufA, bufB, dinv, N, C);
  k_edge_scatter<<<1024, 256, 0, stream>>>(bufA, bufB, src, dst, dinv, E, C);
  k_logsoftmax<<<(N * 32 + 255) / 256, 256, 0, stream>>>(bufB, (float*)d_out, N, C);
}
